// MultiHeadedAttention_77876347011233
// MI455X (gfx1250) — matrix operand order repaired and checked by reading, not compiled, not run on hardware
//
#include <hip/hip_runtime.h>

#define B_  4
#define S_  2048
#define E_  1024
#define H_  16
#define DH_ 64
#define M_  8192
#define N_  1024
#define K_  1024

typedef __attribute__((ext_vector_type(16))) __bf16 bf16x16;
typedef __attribute__((ext_vector_type(8)))  float  f32x8;

union FragAB { bf16x16 v; unsigned u[8]; };
union FragC  { f32x8   v; float    f[8]; };

__device__ inline unsigned short f2bf_u(float f) {
    union { float f; unsigned u; } a; a.f = f;
    unsigned r = a.u + 0x7FFFu + ((a.u >> 16) & 1u);
    return (unsigned short)(r >> 16);
}
__device__ inline unsigned pack2bf(float lo, float hi) {
    return ((unsigned)f2bf_u(hi) << 16) | (unsigned)f2bf_u(lo);
}
__device__ inline float bf2f_u(unsigned short s) {
    union { unsigned u; float f; } a; a.u = ((unsigned)s) << 16; return a.f;
}
// A-fragment (16-bit, 16x32) per-lane K pair base, as dword index
__device__ inline int a_kd(int p, int g) {
    return (p < 4) ? (p + 4 * g) : (8 + (p - 4) + 4 * g);
}
__device__ inline f32x8 wmma_bf16(const FragAB& a, const FragAB& b, f32x8 c) {
    return __builtin_amdgcn_wmma_f32_16x16x32_bf16(false, a.v, false, b.v,
                                                   (short)0, c, false, false);
}

// ---------------- f32 -> bf16 conversion (4 elems/thread) ----------------
__global__ __launch_bounds__(256) void convert_kernel(const float* __restrict__ in,
                                                      unsigned short* __restrict__ out,
                                                      int n4) {
    int i = blockIdx.x * 256 + threadIdx.x;
    if (i < n4) {
        float4 v = *(const float4*)&in[i * 4];
        uint2 o;
        o.x = pack2bf(v.x, v.y);
        o.y = pack2bf(v.z, v.w);
        *(uint2*)&out[i * 4] = o;
    }
}

// ---- GEMM: C[8192,1024] = A[8192,1024](bf16) @ W[1024,1024]^T(bf16) + bias ----
// 128x128 tile/block, 8 waves (2x4), double-buffered LDS, uint4 staging.
#define ASW 20   // LDS row stride in dwords (80B = 5*16B -> uint4-aligned rows)
template<bool OUT_F32>
__global__ __launch_bounds__(256) void gemm_kernel(const unsigned short* __restrict__ A,
                                                   const unsigned short* __restrict__ Wb,
                                                   const float* __restrict__ bias,
                                                   unsigned short* __restrict__ outB,
                                                   float* __restrict__ outF) {
    __shared__ alignas(16) unsigned lds_a[2][128 * ASW];
    __shared__ alignas(16) unsigned lds_w[2][128 * ASW];
    const unsigned* A2 = (const unsigned*)A;
    const unsigned* W2 = (const unsigned*)Wb;
    const int t = threadIdx.x;
    const int wave = t >> 5, lane = t & 31, lane16 = lane & 15, g = lane >> 4;
    const int wm = wave & 1, wn = wave >> 1;
    const int m0 = blockIdx.x * 128, n0 = blockIdx.y * 128;
    const int srow = t >> 2, sq = t & 3;

    uint4 ra[2], rw[2];
    auto load_tile = [&](int kk) {
        #pragma unroll
        for (int i = 0; i < 2; i++) {
            int row = srow + 64 * i;
            ra[i] = ((const uint4*)&A2[(size_t)(m0 + row) * (K_ / 2) + (kk >> 1)])[sq];
            rw[i] = ((const uint4*)&W2[(size_t)(n0 + row) * (K_ / 2) + (kk >> 1)])[sq];
        }
    };
    auto store_tile = [&](int buf) {
        #pragma unroll
        for (int i = 0; i < 2; i++) {
            int row = srow + 64 * i;
            *(uint4*)&lds_a[buf][row * ASW + sq * 4] = ra[i];
            *(uint4*)&lds_w[buf][row * ASW + sq * 4] = rw[i];
        }
    };

    FragC acc[4][2];
    #pragma unroll
    for (int i = 0; i < 4; i++)
        #pragma unroll
        for (int j = 0; j < 2; j++)
            #pragma unroll
            for (int r = 0; r < 8; r++) acc[i][j].f[r] = 0.f;

    load_tile(0);
    store_tile(0);
    const int NT = K_ / 32;
    for (int it = 0; it < NT; it++) {
        __syncthreads();                          // lds[cur] ready
        const int cur = it & 1;
        const bool more = (it + 1 < NT);
        if (more) load_tile((it + 1) * 32);       // overlap global with WMMA below

        FragAB af[4], bfr[2];
        #pragma unroll
        for (int st = 0; st < 4; st++) {
            int row = wm * 64 + st * 16 + lane16;
            #pragma unroll
            for (int p = 0; p < 8; p++)
                af[st].u[p] = lds_a[cur][row * ASW + a_kd(p, g)];
        }
        #pragma unroll
        for (int sn = 0; sn < 2; sn++) {
            int row = wn * 32 + sn * 16 + lane16;
            #pragma unroll
            for (int v = 0; v < 8; v++)
                bfr[sn].u[v] = lds_w[cur][row * ASW + a_kd(v, g)];
        }
        #pragma unroll
        for (int st = 0; st < 4; st++)
            #pragma unroll
            for (int sn = 0; sn < 2; sn++)
                acc[st][sn].v = wmma_bf16(af[st], bfr[sn], acc[st][sn].v);

        if (more) {
            __syncthreads();                      // prev-iter consumers of lds[cur^1] done
            store_tile(cur ^ 1);
        }
    }
    // epilogue: lane holds (M = r + 8g, N = lane16) per C VGPR
    #pragma unroll
    for (int st = 0; st < 4; st++) {
        #pragma unroll
        for (int sn = 0; sn < 2; sn++) {
            int col = n0 + wn * 32 + sn * 16 + lane16;
            float bv = bias[col];
            #pragma unroll
            for (int r = 0; r < 8; r++) {
                int row = m0 + wm * 64 + st * 16 + r + 8 * g;
                float val = acc[st][sn].f[r] + bv;
                if (OUT_F32) outF[(size_t)row * N_ + col] = val;
                else         outB[(size_t)row * N_ + col] = f2bf_u(val);
            }
        }
    }
}

// ---------------- RoPE (in-place on bf16 Q and K) ----------------
__global__ __launch_bounds__(256) void rope_kernel(unsigned short* __restrict__ Q,
                                                   unsigned short* __restrict__ Kt) {
    int n = blockIdx.x * 256 + threadIdx.x;      // n < B*S*H*32
    int j = n & 31;
    int h = (n >> 5) & 15;
    int s = (n >> 9) & (S_ - 1);
    int b = n >> 20;
    float ang = (float)s * __expf(-(float)j * 0.2878231366242557f); // 10000^(-j/32)
    float sn, cs;
    __sincosf(ang, &sn, &cs);
    size_t base = ((size_t)(b * S_ + s)) * E_ + h * DH_;
    float q1 = bf2f_u(Q[base + j]), q2 = bf2f_u(Q[base + j + 32]);
    Q[base + j]      = f2bf_u(q1 * cs - q2 * sn);
    Q[base + j + 32] = f2bf_u(q2 * cs + q1 * sn);
    float k1 = bf2f_u(Kt[base + j]), k2 = bf2f_u(Kt[base + j + 32]);
    Kt[base + j]      = f2bf_u(k1 * cs - k2 * sn);
    Kt[base + j + 32] = f2bf_u(k2 * cs + k1 * sn);
}

// ---------------- flash attention: 128 q-rows/block, 8 waves x 16 rows ----------------
#define KSW 36   // K tile row stride, dwords (144B = 9*16B)
#define VSW 20   // V^T tile row stride, dwords (80B = 5*16B)
#define PSTR 34  // P tile row stride, floats
__global__ __launch_bounds__(256) void attn_kernel(const unsigned short* __restrict__ Q,
                                                   const unsigned short* __restrict__ Kg,
                                                   const unsigned short* __restrict__ Vg,
                                                   unsigned short* __restrict__ Og,
                                                   const int* __restrict__ seqlens) {
    __shared__ alignas(16) unsigned lds_k[32 * KSW];   // [key][d-pairs]
    __shared__ alignas(16) unsigned lds_v[64 * VSW];   // [d][key-pairs] (transposed)
    __shared__ float lds_p[8 * 16 * PSTR];             // per-wave P, f32
    const unsigned* Q2 = (const unsigned*)Q;
    const unsigned* K2 = (const unsigned*)Kg;
    const int t = threadIdx.x;
    const int wave = t >> 5, lane = t & 31, lane16 = lane & 15, g = lane >> 4;
    const int h = blockIdx.y, b = blockIdx.z;
    const int q0 = blockIdx.x * 128;
    const int q0w = q0 + wave * 16;
    const int slen = seqlens[b];

    // resident Q A-fragments (two d-halves of DH=64)
    FragAB qf[2];
    {
        size_t qb = ((size_t)(b * S_ + q0w + lane16)) * (E_ / 2) + h * (DH_ / 2);
        #pragma unroll
        for (int dh = 0; dh < 2; dh++)
            #pragma unroll
            for (int p = 0; p < 8; p++)
                qf[dh].u[p] = Q2[qb + dh * 16 + a_kd(p, g)];
    }

    FragC o[4];
    #pragma unroll
    for (int i = 0; i < 4; i++)
        #pragma unroll
        for (int r = 0; r < 8; r++) o[i].f[r] = 0.f;
    float mrow[8], lrow[8];
    #pragma unroll
    for (int r = 0; r < 8; r++) { mrow[r] = -1e30f; lrow[r] = 0.f; }

    const int nt = q0 / 32 + 4;                  // causal: keys <= q0+127
    float* ldsp = &lds_p[wave * 16 * PSTR];

    for (int it = 0; it < nt; it++) {
        const int kb = it * 32;
        __syncthreads();
        {   // K tile: 32 keys x 32 dwords = 256 uint4, one per thread
            int row = t >> 3, q = t & 7;
            uint4 kv = ((const uint4*)&K2[(size_t)(b * S_ + kb + row) * (E_ / 2)
                                          + h * (DH_ / 2)])[q];
            *(uint4*)&lds_k[row * KSW + q * 4] = kv;
        }
        #pragma unroll
        for (int i = 0; i < 4; i++) {            // V^T tile: d fastest for coalescing
            int c = t + 256 * i;
            int d = c & 63, kp = c >> 6;         // key pair kp -> keys 2kp,2kp+1
            size_t gb = ((size_t)(b * S_ + kb + 2 * kp)) * E_ + h * DH_ + d;
            lds_v[d * VSW + kp] = ((unsigned)Vg[gb + E_] << 16) | (unsigned)Vg[gb];
        }
        if (it + 1 < nt) {                       // CDNA5 prefetch of next tile
            size_t nb = ((size_t)(b * S_ + kb + 32 + (t >> 3))) * E_ + h * DH_ + (t & 7) * 8;
            __builtin_prefetch(&Kg[nb], 0, 0);
            __builtin_prefetch(&Vg[nb], 0, 0);
        }
        __syncthreads();

        // Skip tiles fully above this wave's causal diagonal (contribution is zero).
        // Wave-uniform condition; no barrier inside -> safe to skip.
        if (kb > q0w + 15) continue;

        // scores: S[16x32] = Q(16x64) . K^T(64x32)
        FragC sc[2];
        #pragma unroll
        for (int kg = 0; kg < 2; kg++) {
            FragAB kf0, kf1;
            int krow = kg * 16 + lane16;
            #pragma unroll
            for (int v = 0; v < 8; v++) {
                kf0.u[v] = lds_k[krow * KSW + a_kd(v, g)];
                kf1.u[v] = lds_k[krow * KSW + 16 + a_kd(v, g)];
            }
            f32x8 c = {};
            c = wmma_bf16(qf[0], kf0, c);
            c = wmma_bf16(qf[1], kf1, c);
            sc[kg].v = c;
        }
        // scale (+ causal/len mask only on boundary tiles)
        if ((kb + 31 > q0w) || (kb + 32 > slen)) {
            #pragma unroll
            for (int kg = 0; kg < 2; kg++) {
                int kabs = kb + kg * 16 + lane16;
                #pragma unroll
                for (int r = 0; r < 8; r++) {
                    int qabs = q0w + r + 8 * g;
                    bool ok = (kabs <= qabs) && (kabs < slen);
                    sc[kg].f[r] = ok ? (sc[kg].f[r] * 0.125f) : -1e30f;
                }
            }
        } else {
            #pragma unroll
            for (int kg = 0; kg < 2; kg++)
                #pragma unroll
                for (int r = 0; r < 8; r++) sc[kg].f[r] *= 0.125f;
        }
        // online softmax (rows live in 16-lane groups)
        float alpha[8];
        #pragma unroll
        for (int r = 0; r < 8; r++) {
            float tmax = fmaxf(sc[0].f[r], sc[1].f[r]);
            tmax = fmaxf(tmax, __shfl_xor(tmax, 1));
            tmax = fmaxf(tmax, __shfl_xor(tmax, 2));
            tmax = fmaxf(tmax, __shfl_xor(tmax, 4));
            tmax = fmaxf(tmax, __shfl_xor(tmax, 8));
            float mnew = fmaxf(mrow[r], tmax);
            alpha[r] = __expf(mrow[r] - mnew);
            mrow[r] = mnew;
            float p0 = __expf(sc[0].f[r] - mnew);
            float p1 = __expf(sc[1].f[r] - mnew);
            int prow = r + 8 * g;
            ldsp[prow * PSTR + lane16]      = p0;
            ldsp[prow * PSTR + 16 + lane16] = p1;
            float ps = p0 + p1;
            ps += __shfl_xor(ps, 1);
            ps += __shfl_xor(ps, 2);
            ps += __shfl_xor(ps, 4);
            ps += __shfl_xor(ps, 8);
            lrow[r] = lrow[r] * alpha[r] + ps;
        }
        #pragma unroll
        for (int f = 0; f < 4; f++)
            #pragma unroll
            for (int r = 0; r < 8; r++) o[f].f[r] *= alpha[r];

        asm volatile("" ::: "memory");   // wave-local: DS in-order, block reordering

        // P A-fragment (16x32), f32 pairs -> packed bf16 dwords
        FragAB pf;
        #pragma unroll
        for (int p = 0; p < 8; p++) {
            int kd = a_kd(p, g) * 2;
            pf.u[p] = pack2bf(ldsp[lane16 * PSTR + kd], ldsp[lane16 * PSTR + kd + 1]);
        }
        // O += P(16x32) . V(32x64)
        #pragma unroll
        for (int sn = 0; sn < 4; sn++) {
            FragAB vf;
            int dcol = sn * 16 + lane16;
            #pragma unroll
            for (int v = 0; v < 8; v++)
                vf.u[v] = lds_v[dcol * VSW + a_kd(v, g)];
            o[sn].v = wmma_bf16(pf, vf, o[sn].v);
        }
    }
    // epilogue
    #pragma unroll
    for (int r = 0; r < 8; r++) {
        int qabs = q0w + r + 8 * g;
        float scale = (qabs < slen) ? (1.0f / lrow[r]) : 0.0f;
        size_t obase = ((size_t)(b * S_ + qabs)) * E_ + h * DH_;
        #pragma unroll
        for (int sn = 0; sn < 4; sn++)
            Og[obase + sn * 16 + lane16] = f2bf_u(o[sn].f[r] * scale);
    }
}

extern "C" void kernel_launch(void* const* d_in, const int* in_sizes, int n_in,
                              void* d_out, int out_size, void* d_ws, size_t ws_size,
                              hipStream_t stream) {
    const float* X  = (const float*)d_in[0];
    const float* Wq = (const float*)d_in[1];
    const float* bq = (const float*)d_in[2];
    const float* Wk = (const float*)d_in[3];
    const float* bk = (const float*)d_in[4];
    const float* Wv = (const float*)d_in[5];
    const float* bv = (const float*)d_in[6];
    const float* Wo = (const float*)d_in[7];
    const float* bo = (const float*)d_in[8];
    const int*  seq = (const int*)d_in[9];
    float* out = (float*)d_out;

    const size_t TE = (size_t)B_ * S_ * E_;      // 8,388,608
    const size_t EE = (size_t)E_ * E_;           // 1,048,576
    unsigned short* Xb  = (unsigned short*)d_ws;
    unsigned short* Qb  = Xb + TE;
    unsigned short* Kb  = Xb + 2 * TE;
    unsigned short* Vb  = Xb + 3 * TE;
    unsigned short* Wqb = Xb + 4 * TE;
    unsigned short* Wkb = Wqb + EE;
    unsigned short* Wvb = Wkb + EE;
    unsigned short* Wob = Wvb + EE;
    unsigned short* Ab  = Xb;                    // X dead after QKV projections

    convert_kernel<<<(int)(TE / 1024), 256, 0, stream>>>(X,  Xb,  (int)(TE / 4));
    convert_kernel<<<(int)(EE / 1024), 256, 0, stream>>>(Wq, Wqb, (int)(EE / 4));
    convert_kernel<<<(int)(EE / 1024), 256, 0, stream>>>(Wk, Wkb, (int)(EE / 4));
    convert_kernel<<<(int)(EE / 1024), 256, 0, stream>>>(Wv, Wvb, (int)(EE / 4));
    convert_kernel<<<(int)(EE / 1024), 256, 0, stream>>>(Wo, Wob, (int)(EE / 4));

    dim3 gg(M_ / 128, N_ / 128);                 // 64 x 8
    gemm_kernel<false><<<gg, 256, 0, stream>>>(Xb, Wqb, bq, Qb, nullptr);
    gemm_kernel<false><<<gg, 256, 0, stream>>>(Xb, Wkb, bk, Kb, nullptr);
    gemm_kernel<false><<<gg, 256, 0, stream>>>(Xb, Wvb, bv, Vb, nullptr);

    rope_kernel<<<(B_ * S_ * H_ * 32) / 256, 256, 0, stream>>>(Qb, Kb);

    attn_kernel<<<dim3(S_ / 128, H_, B_), 256, 0, stream>>>(Qb, Kb, Vb, Ab, seq);

    gemm_kernel<true><<<gg, 256, 0, stream>>>(Ab, Wob, bo, nullptr, out);
}
